// MARN_81561428951461
// MI455X (gfx1250) — matrix operand order repaired and checked by reading, not compiled, not run on hardware
//
#include <hip/hip_runtime.h>

typedef __bf16 bf16_t;
typedef __attribute__((ext_vector_type(16))) __bf16 v16bf;
typedef __attribute__((ext_vector_type(8)))  __bf16 v8bf;
typedef __attribute__((ext_vector_type(8)))  float  v8f;

#define T_LEN   256
#define NBLK    16
#define THREADS 256

// ---------- LDS layout for the recurrent kernel (bytes) ----------
#define SM_S      0        // float [16][512] scratch (gates / att-head / final)
#define SM_HZ0    32768    // bf16  [16][256] {h0|h1|h2|z} double buffer A
#define SM_HZ1    40960    // bf16  [16][256] double buffer B
#define SM_HF     49152    // float [16][192] h (f32) for out_tensor
#define SM_C0     61440    // float [16][128]
#define SM_C1     69632    // float [16][32]
#define SM_C2     71680    // float [16][32]
#define SM_ATT    73728    // bf16  [16][768] att_out
#define SM_M1     98304    // bf16  [16][256] attention hidden (relu)
#define SM_MID    106496   // bf16  [16][64]  compression hidden
#define SM_DBUF   108544   // float [16][64]  d-concat before softmax
#define SM_REDA   112640   // float [256]
#define SM_REDB   113664   // float [256]
#define SM_TOTAL  114688   // 112 KB dynamic LDS

// Native hardware f32->bf16 conversion (compiler picks v_cvt on gfx1250)
__device__ __forceinline__ bf16_t f2bf(float f) { return (bf16_t)f; }
__device__ __forceinline__ float sigmoidf_(float x) { return 1.0f / (1.0f + __expf(-x)); }

// ---------- WMMA bf16 16x16x32 fragment helpers (wave32) ----------
// A: 16 x K row-major in LDS. lane holds row M=lane&15; elems 0..7 -> K=kb+e,
// 8..15 -> K=16+kb+e, kb=8*(lane>>4).  Two contiguous 16B runs -> 2x ds_load_b128.
__device__ __forceinline__ v16bf load_a_frag(const bf16_t* A, int lda, int k0, int lane) {
  const bf16_t* p = A + (size_t)(lane & 15) * lda + k0 + ((lane >> 4) << 3);
  union { v16bf v; v8bf h[2]; } u;
  u.h[0] = *(const v8bf*)(p);
  u.h[1] = *(const v8bf*)(p + 16);
  return u.v;
}
// B: pre-packed fragment order [frag][lane][16]; one 32B contiguous load per lane.
__device__ __forceinline__ v16bf load_b_packed(const bf16_t* P, int fragIdx, int lane) {
  return *(const v16bf*)(P + ((size_t)fragIdx * 32 + lane) * 16);
}
__device__ __forceinline__ v8f wmma_bf16(v8f acc, v16bf a, v16bf b) {
  return __builtin_amdgcn_wmma_f32_16x16x32_bf16(false, a, false, b, (short)0, acc, false, false);
}
// acc += A[16 x klen at col ak0] @ W[rows wk0..wk0+klen) for output tile `ntile`,
// where P is W packed with `ksteps` K-steps per N-tile. wk0, klen multiples of 32.
__device__ __forceinline__ v8f wmma_seg_p(v8f acc, const bf16_t* A, int lda, int ak0,
                                          const bf16_t* P, int ksteps, int wk0,
                                          int klen, int ntile, int lane) {
  for (int k = 0; k < klen; k += 32)
    acc = wmma_bf16(acc, load_a_frag(A, lda, ak0 + k, lane),
                         load_b_packed(P, ntile * ksteps + ((wk0 + k) >> 5), lane));
  return acc;
}

// ---------- pack f32 weight [K][N] -> bf16 WMMA-B fragment layout ----------
// P[((ntile*ksteps + kstep)*32 + lane)*16 + e] = W[kstep*32 + 8*(lane>>4) + (e&7) + 16*(e>>3)][ntile*16 + (lane&15)]
__global__ void pack_b_kernel(const float* __restrict__ W, bf16_t* __restrict__ P,
                              int K, int N) {
  const int ksteps = ((K + 31) & ~31) >> 5;
  const int total = (N >> 4) * ksteps * 512;
  int idx = blockIdx.x * 256 + threadIdx.x;
  if (idx >= total) return;
  const int e = idx & 15;
  const int lane = (idx >> 4) & 31;
  const int frag = idx >> 9;
  const int kstep = frag % ksteps;
  const int ntile = frag / ksteps;
  const int k = kstep * 32 + ((lane >> 4) << 3) + (e & 7) + ((e >> 3) << 4);
  const int n = ntile * 16 + (lane & 15);
  P[idx] = f2bf((k < K) ? W[(size_t)k * N + n] : 0.0f);
}

// ---------- Phase 1: pre[t][b][n] = x[b][t][:] @ W + bias  (M = B*T = 65536) ----------
__global__ __launch_bounds__(256) void xw_gemm_kernel(
    const float* __restrict__ X, const bf16_t* __restrict__ Wp,
    const float* __restrict__ bias, float* __restrict__ out,
    int K, int N, int ksteps) {
  __shared__ __align__(16) bf16_t As[64 * 32];
  const int tid = threadIdx.x, lane = tid & 31, wave = tid >> 5;
  const int row0 = blockIdx.x * 64;
  const int mloc = (wave & 3) * 16;
  const int ntile = blockIdx.y * 2 + (wave >> 2);
  v8f acc = {};
  for (int k0 = 0; k0 < K; k0 += 32) {
    for (int e = tid; e < 64 * 32; e += 256) {
      int r = e >> 5, k = e & 31;
      As[e] = (k0 + k < K) ? f2bf(X[(size_t)(row0 + r) * K + k0 + k]) : f2bf(0.0f);
    }
    __syncthreads();
    acc = wmma_bf16(acc, load_a_frag(As + mloc * 32, 32, 0, lane),
                         load_b_packed(Wp, ntile * ksteps + (k0 >> 5), lane));
    __syncthreads();
  }
  const int mtop = 8 * (lane >> 4);
  const int n = ntile * 16 + (lane & 15);
#pragma unroll
  for (int r = 0; r < 8; ++r) {
    int row = row0 + mloc + mtop + r;            // row = b*T + t
    int b = row >> 8, t = row & 255;             // T == 256
    out[((size_t)t * 256 + b) * N + n] = acc[r] + bias[n];  // time-leading layout
  }
}

// ---------- row softmax over LDS buffer [16][width] ----------
__device__ __forceinline__ void row_softmax(float* buf, int width, int stride,
                                            float* redA, float* redB, int tid) {
  const int m = tid >> 4, j16 = tid & 15;
  float mx = -3.4e38f;
  for (int j = j16; j < width; j += 16) mx = fmaxf(mx, buf[m * stride + j]);
  redA[tid] = mx;
  __syncthreads();
  if (j16 == 0) {
    float v = redA[m * 16];
    for (int k = 1; k < 16; ++k) v = fmaxf(v, redA[m * 16 + k]);
    redB[m] = v;
  }
  __syncthreads();
  const float rm = redB[m];
  float s = 0.f;
  for (int j = j16; j < width; j += 16) s += __expf(buf[m * stride + j] - rm);
  redA[tid] = s;
  __syncthreads();
  if (j16 == 0) {
    float v = 0.f;
    for (int k = 0; k < 16; ++k) v += redA[m * 16 + k];
    redB[16 + m] = v;
  }
  __syncthreads();
  const float inv = 1.0f / redB[16 + m];
  for (int j = j16; j < width; j += 16)
    buf[m * stride + j] = __expf(buf[m * stride + j] - rm) * inv;
  __syncthreads();
}

struct RArgs {
  const float *pre0, *pre1, *pre2;
  const bf16_t *U0, *V0, *U1, *V1, *U2, *V2;     // packed
  const bf16_t *aw1, *aw2;                        // packed
  const bf16_t *cw1_0, *cw2_0, *cw1_1, *cw2_1, *cw1_2, *cw2_2;  // packed
  const bf16_t *fw1;                              // packed
  const float *ab1, *ab2, *cb1_0, *cb2_0, *cb1_1, *cb2_1, *cb1_2, *cb2_2, *fb1;
  const float *fw2, *fb2;
  float* out;
};

// ---------- Phase 2: persistent recurrent kernel; 1 WG = 16 batch rows, all T steps ----------
__global__ __launch_bounds__(256, 1) void marn_rnn_kernel(RArgs g) {
  extern __shared__ char smem[];
  float*  S      = (float*)(smem + SM_S);
  bf16_t* hz0    = (bf16_t*)(smem + SM_HZ0);
  bf16_t* hz1    = (bf16_t*)(smem + SM_HZ1);
  float*  hF     = (float*)(smem + SM_HF);
  float*  c0     = (float*)(smem + SM_C0);
  float*  c1     = (float*)(smem + SM_C1);
  float*  c2     = (float*)(smem + SM_C2);
  bf16_t* attout = (bf16_t*)(smem + SM_ATT);
  bf16_t* m1     = (bf16_t*)(smem + SM_M1);
  bf16_t* mid    = (bf16_t*)(smem + SM_MID);
  float*  dbuf   = (float*)(smem + SM_DBUF);
  float*  redA   = (float*)(smem + SM_REDA);
  float*  redB   = (float*)(smem + SM_REDB);

  const int tid = threadIdx.x, lane = tid & 31, wave = tid >> 5;
  const int bbase = blockIdx.x * 16;

  for (int e = tid; e < 16 * 256; e += 256) hz0[e] = f2bf(0.0f);
  for (int e = tid; e < 16 * 192; e += 256) hF[e] = 0.0f;
  for (int e = tid; e < 16 * 128; e += 256) c0[e] = 0.0f;
  for (int e = tid; e < 16 * 32;  e += 256) { c1[e] = 0.0f; c2[e] = 0.0f; }
  __syncthreads();

  bf16_t* hzbuf[2] = { hz0, hz1 };
  int cur = 0;

  const int     HD[3]   = { 128, 32, 32 };
  const int     HOFF[3] = { 0, 128, 160 };
  const int     KSU[3]  = { 4, 1, 1 };            // ksteps of U_i (K = hd)
  const float*  PRE[3]  = { g.pre0, g.pre1, g.pre2 };
  const bf16_t* UU[3]   = { g.U0, g.U1, g.U2 };
  const bf16_t* VV[3]   = { g.V0, g.V1, g.V2 };
  float*        CC[3]   = { c0, c1, c2 };

  for (int t = 0; t < T_LEN; ++t) {
    bf16_t* hz  = hzbuf[cur];
    bf16_t* hzn = hzbuf[cur ^ 1];

    // Prefetch next step's pre-activation slabs (global_prefetch_b8) so the
    // serial scan's next-iteration HBM/L2 traffic overlaps this step's WMMAs.
    if (t + 1 < T_LEN) {
      const char* p0 = (const char*)(PRE[0] + ((size_t)(t + 1) * 256 + bbase) * 512);
      const char* p1 = (const char*)(PRE[1] + ((size_t)(t + 1) * 256 + bbase) * 128);
      const char* p2 = (const char*)(PRE[2] + ((size_t)(t + 1) * 256 + bbase) * 128);
      __builtin_prefetch(p0 + tid * 128, 0, 1);   // 256 threads x 128B = 32KB
      if (tid < 64) {
        __builtin_prefetch(p1 + tid * 128, 0, 1); // 8KB
        __builtin_prefetch(p2 + tid * 128, 0, 1); // 8KB
      }
    }

    // ===== Phase A: three LSTHM cells =====
#pragma unroll
    for (int s = 0; s < 3; ++s) {
      const int hd = HD[s], N = 4 * hd, ntiles = N / 16;
      const float* pre = PRE[s] + ((size_t)t * 256 + bbase) * N;   // [t][b][n]
      for (int tile = wave; tile < ntiles; tile += 8) {
        const int n0 = tile * 16;
        v8f acc = {};
        acc = wmma_seg_p(acc, hz, 256, HOFF[s], UU[s], KSU[s], 0, hd, tile, lane);  // h_i @ U_i
        acc = wmma_seg_p(acc, hz, 256, 192,     VV[s], 2,      0, 64, tile, lane);  // z   @ V_i
        const int mtop = 8 * (lane >> 4);
        const int n = n0 + (lane & 15);
#pragma unroll
        for (int r = 0; r < 8; ++r) {
          const int m = mtop + r;
          S[m * 512 + n] = acc[r] + pre[(size_t)m * N + n];
        }
      }
      __syncthreads();
      for (int e = tid; e < 16 * hd; e += 256) {
        const int m = e / hd, n = e % hd;
        const float f  = sigmoidf_(S[m * 512 + n]);
        const float ii = sigmoidf_(S[m * 512 + hd + n]);
        const float o  = sigmoidf_(S[m * 512 + 2 * hd + n]);
        const float ch = tanhf(S[m * 512 + 3 * hd + n]);
        const float cv = f * CC[s][m * hd + n] + ii * ch;
        const float hv = tanhf(cv) * o;
        CC[s][m * hd + n] = cv;
        hF[m * 192 + HOFF[s] + n] = hv;
        hzn[m * 256 + HOFF[s] + n] = f2bf(hv);
      }
      __syncthreads();
    }

    // ===== Phase B: m1 = relu(hcat @ aw1 + ab1)  [16 x 256] =====
    for (int tile = wave; tile < 16; tile += 8) {
      const int n0 = tile * 16;
      v8f acc = {};
      acc = wmma_seg_p(acc, hzn, 256, 0, g.aw1, 6, 0, 192, tile, lane);
      const int mtop = 8 * (lane >> 4);
      const int n = n0 + (lane & 15);
#pragma unroll
      for (int r = 0; r < 8; ++r) {
        float v = acc[r] + g.ab1[n];
        m1[(mtop + r) * 256 + n] = f2bf(v > 0.f ? v : 0.f);
      }
    }
    __syncthreads();

    // ===== Phase C: per-head att2 -> sigmoid -> softmax(192) -> * out_tensor =====
    for (int a = 0; a < 4; ++a) {
      for (int tile = wave; tile < 12; tile += 8) {
        const int n0 = tile * 16;
        v8f acc = {};
        acc = wmma_seg_p(acc, m1, 256, 0, g.aw2, 8, 0, 256, a * 12 + tile, lane);
        const int mtop = 8 * (lane >> 4);
        const int n = n0 + (lane & 15);
#pragma unroll
        for (int r = 0; r < 8; ++r)
          S[(mtop + r) * 192 + n] = sigmoidf_(acc[r] + g.ab2[a * 192 + n]);
      }
      __syncthreads();
      row_softmax(S, 192, 192, redA, redB, tid);
      for (int e = tid; e < 16 * 192; e += 256) {
        const int m = e / 192, j = e % 192;
        const int u = a * 192 + j;
        float hval;
        if (u < 512)      hval = hF[m * 192 + (u & 127)];
        else if (u < 640) hval = hF[m * 192 + 128 + ((u - 512) & 31)];
        else              hval = hF[m * 192 + 160 + ((u - 640) & 31)];
        attout[m * 768 + u] = f2bf(S[m * 192 + j] * hval);
      }
      __syncthreads();
    }

    // ===== Phase D: compression nets -> dbuf[16 x 64] =====
    // d0
    for (int tile = wave; tile < 4; tile += 8) {
      const int n0 = tile * 16;
      v8f acc = {};
      acc = wmma_seg_p(acc, attout, 768, 0, g.cw1_0, 16, 0, 512, tile, lane);
      const int mtop = 8 * (lane >> 4); const int n = n0 + (lane & 15);
#pragma unroll
      for (int r = 0; r < 8; ++r) {
        float v = acc[r] + g.cb1_0[n];
        mid[(mtop + r) * 64 + n] = f2bf(v > 0.f ? v : 0.f);
      }
    }
    __syncthreads();
    for (int tile = wave; tile < 2; tile += 8) {
      const int n0 = tile * 16;
      v8f acc = {};
      acc = wmma_seg_p(acc, mid, 64, 0, g.cw2_0, 2, 0, 64, tile, lane);
      const int mtop = 8 * (lane >> 4); const int n = n0 + (lane & 15);
#pragma unroll
      for (int r = 0; r < 8; ++r)
        dbuf[(mtop + r) * 64 + n] = sigmoidf_(acc[r] + g.cb2_0[n]);
    }
    __syncthreads();
    // d1
    for (int tile = wave; tile < 2; tile += 8) {
      const int n0 = tile * 16;
      v8f acc = {};
      acc = wmma_seg_p(acc, attout, 768, 512, g.cw1_1, 4, 0, 128, tile, lane);
      const int mtop = 8 * (lane >> 4); const int n = n0 + (lane & 15);
#pragma unroll
      for (int r = 0; r < 8; ++r) {
        float v = acc[r] + g.cb1_1[n];
        mid[(mtop + r) * 32 + n] = f2bf(v > 0.f ? v : 0.f);
      }
    }
    __syncthreads();
    for (int tile = wave; tile < 1; tile += 8) {
      v8f acc = {};
      acc = wmma_seg_p(acc, mid, 32, 0, g.cw2_1, 1, 0, 32, 0, lane);
      const int mtop = 8 * (lane >> 4); const int n = lane & 15;
#pragma unroll
      for (int r = 0; r < 8; ++r)
        dbuf[(mtop + r) * 64 + 32 + n] = sigmoidf_(acc[r] + g.cb2_1[n]);
    }
    __syncthreads();
    // d2
    for (int tile = wave; tile < 2; tile += 8) {
      const int n0 = tile * 16;
      v8f acc = {};
      acc = wmma_seg_p(acc, attout, 768, 640, g.cw1_2, 4, 0, 128, tile, lane);
      const int mtop = 8 * (lane >> 4); const int n = n0 + (lane & 15);
#pragma unroll
      for (int r = 0; r < 8; ++r) {
        float v = acc[r] + g.cb1_2[n];
        mid[(mtop + r) * 32 + n] = f2bf(v > 0.f ? v : 0.f);
      }
    }
    __syncthreads();
    for (int tile = wave; tile < 1; tile += 8) {
      v8f acc = {};
      acc = wmma_seg_p(acc, mid, 32, 0, g.cw2_2, 1, 0, 32, 0, lane);
      const int mtop = 8 * (lane >> 4); const int n = lane & 15;
#pragma unroll
      for (int r = 0; r < 8; ++r)
        dbuf[(mtop + r) * 64 + 48 + n] = sigmoidf_(acc[r] + g.cb2_2[n]);
    }
    __syncthreads();

    // ===== Phase E: z = softmax(dbuf, 64) =====
    row_softmax(dbuf, 64, 64, redA, redB, tid);
    for (int e = tid; e < 16 * 64; e += 256) {
      const int m = e >> 6, n = e & 63;
      hzn[m * 256 + 192 + n] = f2bf(dbuf[e]);
    }
    __syncthreads();
    cur ^= 1;
  }

  // ===== Final head: out = relu([z|h] @ fw1 + fb1) @ fw2 + fb2 =====
  bf16_t* hz = hzbuf[cur];
  for (int tile = wave; tile < 4; tile += 8) {
    const int n0 = tile * 16;
    v8f acc = {};
    acc = wmma_seg_p(acc, hz, 256, 192, g.fw1, 8, 0,  64,  tile, lane);  // z part (rows 0..63)
    acc = wmma_seg_p(acc, hz, 256, 0,   g.fw1, 8, 64, 192, tile, lane);  // h part (rows 64..255)
    const int mtop = 8 * (lane >> 4); const int n = n0 + (lane & 15);
#pragma unroll
    for (int r = 0; r < 8; ++r) {
      float v = acc[r] + g.fb1[n];
      S[(mtop + r) * 64 + n] = v > 0.f ? v : 0.f;
    }
  }
  __syncthreads();
  if (tid < 16) {
    float s = 0.f;
    for (int n = 0; n < 64; ++n) s += S[tid * 64 + n] * g.fw2[n];
    g.out[bbase + tid] = s + g.fb2[0];
  }
}

extern "C" void kernel_launch(void* const* d_in, const int* in_sizes, int n_in,
                              void* d_out, int out_size, void* d_ws, size_t ws_size,
                              hipStream_t stream) {
  (void)in_sizes; (void)n_in; (void)out_size; (void)ws_size;
  const float* x0 = (const float*)d_in[0];
  const float* x1 = (const float*)d_in[1];
  const float* x2 = (const float*)d_in[2];
  const float* W0 = (const float*)d_in[3];
  const float* U0 = (const float*)d_in[4];
  const float* V0 = (const float*)d_in[5];
  const float* b0 = (const float*)d_in[6];
  const float* W1 = (const float*)d_in[7];
  const float* U1 = (const float*)d_in[8];
  const float* V1 = (const float*)d_in[9];
  const float* b1 = (const float*)d_in[10];
  const float* W2 = (const float*)d_in[11];
  const float* U2 = (const float*)d_in[12];
  const float* V2 = (const float*)d_in[13];
  const float* b2 = (const float*)d_in[14];
  const float* aw1 = (const float*)d_in[15];
  const float* ab1 = (const float*)d_in[16];
  const float* aw2 = (const float*)d_in[17];
  const float* ab2 = (const float*)d_in[18];
  const float* cw1_0 = (const float*)d_in[19];
  const float* cb1_0 = (const float*)d_in[20];
  const float* cw2_0 = (const float*)d_in[21];
  const float* cb2_0 = (const float*)d_in[22];
  const float* cw1_1 = (const float*)d_in[23];
  const float* cb1_1 = (const float*)d_in[24];
  const float* cw2_1 = (const float*)d_in[25];
  const float* cb2_1 = (const float*)d_in[26];
  const float* cw1_2 = (const float*)d_in[27];
  const float* cb1_2 = (const float*)d_in[28];
  const float* cw2_2 = (const float*)d_in[29];
  const float* cb2_2 = (const float*)d_in[30];
  const float* fw1 = (const float*)d_in[31];
  const float* fb1 = (const float*)d_in[32];
  const float* fw2 = (const float*)d_in[33];
  const float* fb2 = (const float*)d_in[34];

  // Workspace: pre0 [T][B][512] | pre1 [T][B][128] | pre2 [T][B][128] | packed bf16 weights
  char* ws = (char*)d_ws;
  float* pre0 = (float*)(ws);
  float* pre1 = (float*)(ws + (size_t)134217728);
  float* pre2 = (float*)(ws + (size_t)167772160);
  char* wb = ws + (size_t)201326592;
  size_t off = 0;
  auto abf = [&](int elems) {
    bf16_t* p = (bf16_t*)(wb + off);
    off += (((size_t)elems * 2) + 255) & ~(size_t)255;
    return p;
  };
  // packed size = ntiles * ksteps * 512 elements
  bf16_t* W0p = abf(32 * 10 * 512);
  bf16_t* W1p = abf(8 * 3 * 512);
  bf16_t* W2p = abf(8 * 2 * 512);
  bf16_t* U0p = abf(32 * 4 * 512);
  bf16_t* V0p = abf(32 * 2 * 512);
  bf16_t* U1p = abf(8 * 1 * 512);
  bf16_t* V1p = abf(8 * 2 * 512);
  bf16_t* U2p = abf(8 * 1 * 512);
  bf16_t* V2p = abf(8 * 2 * 512);
  bf16_t* aw1p = abf(16 * 6 * 512);
  bf16_t* aw2p = abf(48 * 8 * 512);
  bf16_t* cw1_0p = abf(4 * 16 * 512);
  bf16_t* cw2_0p = abf(2 * 2 * 512);
  bf16_t* cw1_1p = abf(2 * 4 * 512);
  bf16_t* cw2_1p = abf(1 * 1 * 512);
  bf16_t* cw1_2p = abf(2 * 4 * 512);
  bf16_t* cw2_2p = abf(1 * 1 * 512);
  bf16_t* fw1p = abf(4 * 8 * 512);

  auto pack = [&](const float* w, bf16_t* p, int K, int N) {
    int total = (N >> 4) * (((K + 31) & ~31) >> 5) * 512;
    pack_b_kernel<<<(total + 255) / 256, 256, 0, stream>>>(w, p, K, N);
  };
  pack(W0, W0p, 300, 512);  pack(W1, W1p, 74, 128);   pack(W2, W2p, 35, 128);
  pack(U0, U0p, 128, 512);  pack(V0, V0p, 64, 512);
  pack(U1, U1p, 32, 128);   pack(V1, V1p, 64, 128);
  pack(U2, U2p, 32, 128);   pack(V2, V2p, 64, 128);
  pack(aw1, aw1p, 192, 256); pack(aw2, aw2p, 256, 768);
  pack(cw1_0, cw1_0p, 512, 64); pack(cw2_0, cw2_0p, 64, 32);
  pack(cw1_1, cw1_1p, 128, 32); pack(cw2_1, cw2_1p, 32, 16);
  pack(cw1_2, cw1_2p, 128, 32); pack(cw2_2, cw2_2p, 32, 16);
  pack(fw1, fw1p, 256, 64);

  // Phase 1: time-independent x@W+b over M = B*T = 65536 rows
  xw_gemm_kernel<<<dim3(1024, 16), 256, 0, stream>>>(x0, W0p, b0, pre0, 300, 512, 10);
  xw_gemm_kernel<<<dim3(1024, 4),  256, 0, stream>>>(x1, W1p, b1, pre1, 74, 128, 3);
  xw_gemm_kernel<<<dim3(1024, 4),  256, 0, stream>>>(x2, W2p, b2, pre2, 35, 128, 2);

  // Phase 2: persistent recurrent scan, 16 WGs x 16 batch rows
  RArgs a;
  a.pre0 = pre0; a.pre1 = pre1; a.pre2 = pre2;
  a.U0 = U0p; a.V0 = V0p; a.U1 = U1p; a.V1 = V1p; a.U2 = U2p; a.V2 = V2p;
  a.aw1 = aw1p; a.aw2 = aw2p;
  a.cw1_0 = cw1_0p; a.cw2_0 = cw2_0p; a.cw1_1 = cw1_1p; a.cw2_1 = cw2_1p;
  a.cw1_2 = cw1_2p; a.cw2_2 = cw2_2p;
  a.fw1 = fw1p;
  a.ab1 = ab1; a.ab2 = ab2;
  a.cb1_0 = cb1_0; a.cb2_0 = cb2_0; a.cb1_1 = cb1_1; a.cb2_1 = cb2_1;
  a.cb1_2 = cb1_2; a.cb2_2 = cb2_2;
  a.fb1 = fb1; a.fw2 = fw2; a.fb2 = fb2;
  a.out = (float*)d_out;

  (void)hipFuncSetAttribute(reinterpret_cast<const void*>(marn_rnn_kernel),
                            hipFuncAttributeMaxDynamicSharedMemorySize, SM_TOTAL);
  marn_rnn_kernel<<<NBLK, THREADS, SM_TOTAL, stream>>>(a);
}
